// RPECausalAttention_67388036874290
// MI455X (gfx1250) — matrix operand order repaired and checked by reading, not compiled, not run on hardware
//
#include <hip/hip_runtime.h>

// ---------------------------------------------------------------------------
// Problem constants (match reference)
// ---------------------------------------------------------------------------
#define D_MODEL 1024
#define N_HEAD  16
#define HD      64
#define BATCH   4
#define SEQ     2048
#define ROWS    (BATCH * SEQ)          // 8192
#define QKV_N   (3 * D_MODEL)          // 3072

typedef __attribute__((ext_vector_type(16))) __bf16 bf16x16;
typedef __attribute__((ext_vector_type(8)))  float  f32x8;
typedef __attribute__((ext_vector_type(4)))  unsigned int u32x4;
typedef __attribute__((ext_vector_type(8)))  int  i32x8;
typedef __attribute__((ext_vector_type(4)))  int  i32x4;

union FragU { bf16x16 v; uint4 u[2]; };

// fp32 -> bf16 (round to nearest even), stored as raw u16
__device__ __forceinline__ unsigned short f2bf(float f) {
    unsigned u = __builtin_bit_cast(unsigned, f);
    u += 0x7FFFu + ((u >> 16) & 1u);
    return (unsigned short)(u >> 16);
}

// Low 32 bits of a generic shared pointer = LDS byte offset (aperture lives
// in the upper 32 bits).
__device__ __forceinline__ unsigned lds_off(const void* p) {
    return (unsigned)(unsigned long long)p;
}

// A-matrix fragment, 16(M) x 32(K) bf16.  ISA layout (CDNA5 ISA):
// lane L: M = L%16; half = L/16; VGPR 0..3 hold K = half*8 + 0..7,
// VGPR 4..7 hold K = 16 + half*8 + 0..7  -> two 16-byte loads per lane.
__device__ __forceinline__ bf16x16 load_frag_a(const unsigned short* p, int rs, int kc) {
    const int lane = threadIdx.x & 31;
    const int m = lane & 15, h8 = (lane >> 4) * 8;
    FragU f;
    f.u[0] = *(const uint4*)(p + m * rs + kc + h8);
    f.u[1] = *(const uint4*)(p + m * rs + kc + 16 + h8);
    return f.v;
}

// B-matrix fragment from a transposed tile stored [N][K] row-major.
// lane L: N = L%16; K = (L/16)*8 + 0..7, then 16 + (L/16)*8 + 0..7 -> two 16-byte loads.
__device__ __forceinline__ bf16x16 load_frag_bt(const unsigned short* p, int rs, int kc) {
    const int lane = threadIdx.x & 31;
    const int n = lane & 15, h8 = (lane >> 4) * 8;
    FragU f;
    f.u[0] = *(const uint4*)(p + n * rs + kc + h8);
    f.u[1] = *(const uint4*)(p + n * rs + kc + 16 + h8);
    return f.v;
}

__device__ __forceinline__ f32x8 wmma_bf16(bf16x16 a, bf16x16 b, f32x8 c) {
    return __builtin_amdgcn_wmma_f32_16x16x32_bf16(
        /*neg_a=*/false, a, /*neg_b=*/false, b,
        /*c_mod=*/(short)0, c, /*reuse_a=*/false, /*reuse_b=*/false);
}

// Per-thread async copy of 16 bytes global -> LDS (ASYNCcnt path).
__device__ __forceinline__ void async_copy_b128(unsigned ldsoff, const void* g) {
    unsigned long long ga = (unsigned long long)g;
    asm volatile("global_load_async_to_lds_b128 %0, %1, off"
                 :: "v"(ldsoff), "v"(ga) : "memory");
}
__device__ __forceinline__ void wait_async0() {
    asm volatile("s_wait_asynccnt 0x0" ::: "memory");
}

// TDM: contiguous 1-D copy of `nelem` bf16 elements global -> LDS.
// D# built per CDNA5 ISA (count=1, type=2, data_size=2B,
// tile_dim0 = nelem, tensor_dim0 = nelem, groups 2/3 zero => <=2D).
// This toolchain's builtin is the 6-arg form:
//   (uint32x4 g0, int32x8 g1, int32x4 g2, int32x4 g3, int32x8 pad, i32 cpol)
__device__ __forceinline__ void tdm_load_1d(unsigned ldsoff, const void* g, unsigned nelem) {
    unsigned long long ga = (unsigned long long)g;
    u32x4 g0;
    g0[0] = 1u;                                      // count=1, user mode
    g0[1] = ldsoff;                                  // lds_addr
    g0[2] = (unsigned)(ga & 0xFFFFFFFFu);            // global_addr[31:0]
    g0[3] = (unsigned)((ga >> 32) & 0x1FFFFFFu) | (2u << 30);  // addr[56:32] | type=2
    i32x8 g1;
    g1[0] = (int)(1u << 16);                         // data_size=1 (2 bytes)
    g1[1] = (int)((nelem & 0xFFFFu) << 16);          // tensor_dim0[15:0] @ bit48
    g1[2] = (int)(((nelem >> 16) & 0xFFFFu) | (1u << 16)); // dim0[31:16] | tensor_dim1=1
    g1[3] = (int)((nelem & 0xFFFFu) << 16);          // tile_dim0 @ bit112
    g1[4] = 0;                                       // tile_dim1=0, tile_dim2=0 (1-D)
    g1[5] = (int)nelem;                              // tensor_dim0_stride lo
    g1[6] = 0;
    g1[7] = 0;
    i32x4 z4 = {0, 0, 0, 0};
    i32x8 z8 = {0, 0, 0, 0, 0, 0, 0, 0};
    __builtin_amdgcn_tensor_load_to_lds(g0, g1, z4, z4, z8, 0);
}

// ---------------------------------------------------------------------------
// Kernel 1: fp32 -> bf16 conversion of x, Wqkv, Wproj
// ---------------------------------------------------------------------------
__global__ __launch_bounds__(256) void cvt_kernel(
    const float* __restrict__ x, const float* __restrict__ wqkv,
    const float* __restrict__ wproj,
    unsigned short* __restrict__ xb, unsigned short* __restrict__ wqb,
    unsigned short* __restrict__ wpb, int nx, int nq, int np) {
    int idx = blockIdx.x * 256 + threadIdx.x;
    if (idx < nx)                xb[idx]            = f2bf(x[idx]);
    else if (idx < nx + nq)      wqb[idx - nx]      = f2bf(wqkv[idx - nx]);
    else if (idx < nx + nq + np) wpb[idx - nx - nq] = f2bf(wproj[idx - nx - nq]);
}

// ---------------------------------------------------------------------------
// Kernel 2: QKV GEMM  [8192,1024] x [1024,3072] (bf16, f32 acc)
// ---------------------------------------------------------------------------
__global__ __launch_bounds__(256) void qkv_gemm(
    const unsigned short* __restrict__ xb, const unsigned short* __restrict__ wb,
    unsigned short* __restrict__ qws, unsigned short* __restrict__ kws,
    unsigned short* __restrict__ vtws) {
    __shared__ unsigned short As[64 * 32];   // [m][k]
    __shared__ unsigned short Bts[64 * 32];  // [n][k] (transposed W tile)
    const int tid = threadIdx.x;
    const int n0 = blockIdx.x * 64, m0 = blockIdx.y * 64;
    const int w = tid >> 5, lane = tid & 31;
    const int wm = w >> 1, wn = w & 1;
    f32x8 acc0 = {}, acc1 = {};

    for (int kc = 0; kc < D_MODEL; kc += 32) {
        __syncthreads();
        {   // A tile: 64x32 = 2048 elems, 8 contiguous per thread (uint4)
            int row = tid >> 2, col = (tid & 3) * 8;
            *(uint4*)(As + row * 32 + col) =
                *(const uint4*)(xb + (m0 + row) * D_MODEL + kc + col);
        }
        // B tile transposed: coalesced global read along N, strided LDS write
        for (int i = 0; i < 8; ++i) {
            int e = tid + i * 256;
            int n = e & 63, k = e >> 6;
            Bts[n * 32 + k] = wb[(kc + k) * QKV_N + n0 + n];
        }
        __syncthreads();
        bf16x16 a  = load_frag_a(As + wm * 16 * 32, 32, 0);
        bf16x16 b0 = load_frag_bt(Bts + (wn * 32) * 32, 32, 0);
        bf16x16 b1 = load_frag_bt(Bts + (wn * 32 + 16) * 32, 32, 0);
        acc0 = wmma_bf16(a, b0, acc0);
        acc1 = wmma_bf16(a, b1, acc1);
    }

    const int half = lane >> 4, N = lane & 15;
    for (int nt = 0; nt < 2; ++nt) {
        f32x8 acc = nt ? acc1 : acc0;
        for (int r = 0; r < 8; ++r) {
            int row = m0 + wm * 16 + r + half * 8;     // global token row
            int c   = n0 + wn * 32 + nt * 16 + N;      // column in [0,3072)
            int b = row >> 11, t = row & (SEQ - 1);
            int which = c >> 10, c2 = c & (D_MODEL - 1);
            int h = c2 >> 6, d = c2 & (HD - 1);
            int bh = b * N_HEAD + h;
            float v = acc[r];
            if (which == 0)      qws[(bh * SEQ + t) * HD + d] = f2bf(v * 0.125f);
            else if (which == 1) kws[(bh * SEQ + t) * HD + d] = f2bf(v);
            else                 vtws[(bh * HD + d) * SEQ + t] = f2bf(v);
        }
    }
}

// ---------------------------------------------------------------------------
// Kernel 3: flash attention with relative-position bias, causal.
// K tile staged by the Tensor Data Mover (TENSORcnt), V^T tile by
// per-thread async global->LDS b128 copies (ASYNCcnt).
// ---------------------------------------------------------------------------
__global__ __launch_bounds__(256) void attn_kernel(
    const unsigned short* __restrict__ qws, const unsigned short* __restrict__ kws,
    const unsigned short* __restrict__ vtws, const float* __restrict__ rel_bias,
    unsigned short* __restrict__ yb) {
    __shared__ unsigned short Ks[32 * 64];        // [token][feat]  (== B^T for QK^T)
    __shared__ unsigned short Vts[64 * 32];       // [feat][token]  (== B^T for P*V)
    __shared__ unsigned short Ps[8 * 16 * 32];    // per-wave P tile [16][32]

    const int tid = threadIdx.x, w = tid >> 5, lane = tid & 31;
    const int h = blockIdx.y, b = blockIdx.z;
    const int bh = b * N_HEAD + h;
    const int qbase = blockIdx.x * 128 + w * 16;
    const int half = lane >> 4, N = lane & 15;

    const unsigned short* qp = qws + (bh * SEQ + qbase) * HD;
    bf16x16 aq0 = load_frag_a(qp, HD, 0);
    bf16x16 aq1 = load_frag_a(qp, HD, 32);

    f32x8 acc[4] = {{}, {}, {}, {}};
    float mrow[8], lrow[8];
    for (int r = 0; r < 8; ++r) { mrow[r] = -1e30f; lrow[r] = 0.0f; }

    const unsigned ldsK = lds_off(Ks);
    const int vr = tid >> 2, vc = (tid & 3) * 8;
    const unsigned ldsV = lds_off(Vts + vr * 32 + vc);

    const int nkv = blockIdx.x * 4 + 4;           // causal kv-block bound
    for (int j = 0; j < nkv; ++j) {
        const int kb = j * 32;
        __syncthreads();
        // K tile: one contiguous 4KB TDM transfer, issued by wave 0
        if (w == 0)
            tdm_load_1d(ldsK, kws + (bh * SEQ + kb) * HD, 32 * HD);
        // V^T tile: 256 x 16B async copies
        async_copy_b128(ldsV, vtws + (bh * HD + vr) * SEQ + kb + vc);
        wait_async0();
        if (w == 0)
            __builtin_amdgcn_s_wait_tensorcnt(0);
        __syncthreads();

        // S = (Q * 1/sqrt(hd)) K^T : two 16-wide kv sub-tiles, K-dim 64
        f32x8 s0 = {}, s1 = {};
        s0 = wmma_bf16(aq0, load_frag_bt(Ks, 64, 0), s0);
        s0 = wmma_bf16(aq1, load_frag_bt(Ks, 64, 32), s0);
        s1 = wmma_bf16(aq0, load_frag_bt(Ks + 16 * 64, 64, 0), s1);
        s1 = wmma_bf16(aq1, load_frag_bt(Ks + 16 * 64, 64, 32), s1);

        // bias + causal mask + row max (rows live in 16-lane halves)
        float rm[8];
        for (int r = 0; r < 8; ++r) {
            int q = qbase + r + half * 8;
            int k0 = kb + N, k1 = kb + 16 + N;
            if (k0 > q) s0[r] = -1e30f; else s0[r] += rel_bias[(q - k0) * N_HEAD + h];
            if (k1 > q) s1[r] = -1e30f; else s1[r] += rel_bias[(q - k1) * N_HEAD + h];
            rm[r] = fmaxf(s0[r], s1[r]);
        }
        for (int off = 1; off < 16; off <<= 1)
            for (int r = 0; r < 8; ++r)
                rm[r] = fmaxf(rm[r], __shfl_xor(rm[r], off, 32));

        float corr[8];
        for (int r = 0; r < 8; ++r) {
            float mn = fmaxf(mrow[r], rm[r]);
            corr[r] = __expf(mrow[r] - mn);
            mrow[r] = mn;
            s0[r] = __expf(s0[r] - mn);
            s1[r] = __expf(s1[r] - mn);
        }
        float rs[8];
        for (int r = 0; r < 8; ++r) rs[r] = s0[r] + s1[r];
        for (int off = 1; off < 16; off <<= 1)
            for (int r = 0; r < 8; ++r)
                rs[r] += __shfl_xor(rs[r], off, 32);
        for (int r = 0; r < 8; ++r) lrow[r] = lrow[r] * corr[r] + rs[r];
        for (int t = 0; t < 4; ++t)
            for (int r = 0; r < 8; ++r) acc[t][r] *= corr[r];

        // C-layout -> A-layout for P via wave-private LDS patch
        unsigned short* pw = Ps + w * (16 * 32);
        for (int r = 0; r < 8; ++r) {
            int M = r + half * 8;
            pw[M * 32 + N]      = f2bf(s0[r]);
            pw[M * 32 + 16 + N] = f2bf(s1[r]);
        }
        __syncthreads();

        bf16x16 ap = load_frag_a(pw, 32, 0);
        for (int t = 0; t < 4; ++t)
            acc[t] = wmma_bf16(ap, load_frag_bt(Vts + t * 16 * 32, 32, 0), acc[t]);
    }

    // epilogue: y[b,t,h*64+d] = acc / l   (bf16 for the out-proj GEMM)
    for (int t = 0; t < 4; ++t)
        for (int r = 0; r < 8; ++r) {
            int tq = qbase + r + half * 8;
            int c  = h * HD + t * 16 + N;
            yb[(b * SEQ + tq) * D_MODEL + c] = f2bf(acc[t][r] / lrow[r]);
        }
}

// ---------------------------------------------------------------------------
// Kernel 4: out projection [8192,1024] x [1024,1024] -> fp32
// ---------------------------------------------------------------------------
__global__ __launch_bounds__(256) void proj_gemm(
    const unsigned short* __restrict__ yb, const unsigned short* __restrict__ wb,
    float* __restrict__ out) {
    __shared__ unsigned short As[64 * 32];
    __shared__ unsigned short Bts[64 * 32];
    const int tid = threadIdx.x;
    const int n0 = blockIdx.x * 64, m0 = blockIdx.y * 64;
    const int w = tid >> 5, lane = tid & 31;
    const int wm = w >> 1, wn = w & 1;
    f32x8 acc0 = {}, acc1 = {};

    for (int kc = 0; kc < D_MODEL; kc += 32) {
        __syncthreads();
        {
            int row = tid >> 2, col = (tid & 3) * 8;
            *(uint4*)(As + row * 32 + col) =
                *(const uint4*)(yb + (m0 + row) * D_MODEL + kc + col);
        }
        for (int i = 0; i < 8; ++i) {
            int e = tid + i * 256;
            int n = e & 63, k = e >> 6;
            Bts[n * 32 + k] = wb[(kc + k) * D_MODEL + n0 + n];
        }
        __syncthreads();
        bf16x16 a  = load_frag_a(As + wm * 16 * 32, 32, 0);
        bf16x16 b0 = load_frag_bt(Bts + (wn * 32) * 32, 32, 0);
        bf16x16 b1 = load_frag_bt(Bts + (wn * 32 + 16) * 32, 32, 0);
        acc0 = wmma_bf16(a, b0, acc0);
        acc1 = wmma_bf16(a, b1, acc1);
    }

    const int half = lane >> 4, N = lane & 15;
    for (int nt = 0; nt < 2; ++nt) {
        f32x8 acc = nt ? acc1 : acc0;
        for (int r = 0; r < 8; ++r) {
            int row = m0 + wm * 16 + r + half * 8;
            int c   = n0 + wn * 32 + nt * 16 + N;
            out[row * D_MODEL + c] = acc[r];
        }
    }
}

// ---------------------------------------------------------------------------
// Host launcher
// ---------------------------------------------------------------------------
extern "C" void kernel_launch(void* const* d_in, const int* in_sizes, int n_in,
                              void* d_out, int out_size, void* d_ws, size_t ws_size,
                              hipStream_t stream) {
    const float* x        = (const float*)d_in[0];   // [B,T,C]
    const float* wqkv     = (const float*)d_in[1];   // [C,3C]
    const float* wproj    = (const float*)d_in[2];   // [C,C]
    const float* rel_bias = (const float*)d_in[3];   // [MAX_LEN,H]
    float* out = (float*)d_out;

    // workspace layout (bytes)
    char* ws = (char*)d_ws;
    const size_t NX = (size_t)ROWS * D_MODEL;        // 8388608
    const size_t NQ = (size_t)D_MODEL * QKV_N;       // 3145728
    const size_t NP = (size_t)D_MODEL * D_MODEL;     // 1048576
    unsigned short* xb   = (unsigned short*)(ws);
    unsigned short* wqb  = (unsigned short*)(ws + 2 * NX);
    unsigned short* wpb  = (unsigned short*)(ws + 2 * (NX + NQ));
    unsigned short* qws  = (unsigned short*)(ws + 2 * (NX + NQ + NP));
    unsigned short* kws  = (unsigned short*)(ws + 2 * (NX + NQ + NP) + 2 * NX);
    unsigned short* vtws = (unsigned short*)(ws + 2 * (NX + NQ + NP) + 4 * NX);
    unsigned short* ybf  = (unsigned short*)(ws + 2 * (NX + NQ + NP) + 6 * NX);

    const int ntot = (int)(NX + NQ + NP);
    cvt_kernel<<<dim3((ntot + 255) / 256), dim3(256), 0, stream>>>(
        x, wqkv, wproj, xb, wqb, wpb, (int)NX, (int)NQ, (int)NP);

    qkv_gemm<<<dim3(QKV_N / 64, ROWS / 64), dim3(256), 0, stream>>>(
        xb, wqb, qws, kws, vtws);

    attn_kernel<<<dim3(SEQ / 128, N_HEAD, BATCH), dim3(256), 0, stream>>>(
        qws, kws, vtws, rel_bias, ybf);

    proj_gemm<<<dim3(D_MODEL / 64, ROWS / 64), dim3(256), 0, stream>>>(
        ybf, wpb, out);
}
